// DKD_12816182411600
// MI455X (gfx1250) — matrix operand order repaired and checked by reading, not compiled, not run on hardware
//
#include <hip/hip_runtime.h>
#include <hip/hip_bf16.h>
#include <math.h>

// ---------------------------------------------------------------------------
// DKD keypoint pipeline for gfx1250 (MI455X, wave32, WMMA, async-LDS)
//   in : scores_map [8,1,1024,1024] f32
//   out: keypoints_xy [8,4096,2] || scoredispersity [8,4096] || kptscores [8,4096]
// ---------------------------------------------------------------------------

#define IMG_B 8
#define IMG_H 1024
#define IMG_W 1024
#define IMG_N (IMG_H * IMG_W)
#define TOPK  4096
#define NBUCKET 16384           // bits>>16 of f32 in (0,1) -> max bucket 0x3F7F
#define CAP2  8192              // candidates per image after threshold (pow2)

typedef __attribute__((ext_vector_type(16))) _Float16 v16h;
typedef __attribute__((ext_vector_type(8)))  float    v8f;

#if defined(__has_builtin)
#if __has_builtin(__builtin_amdgcn_global_load_async_to_lds_b32) && \
    __has_builtin(__builtin_amdgcn_s_wait_asynccnt)
#define USE_ASYNC_LDS 1
#endif
#endif

#if USE_ASYNC_LDS
typedef __attribute__((address_space(1))) int g_int;   // global
typedef __attribute__((address_space(3))) int l_int;   // LDS
#endif

__device__ __forceinline__ float max5(const float* a, int stride, int ci) {
    float m = -INFINITY;
#pragma unroll
    for (int di = -2; di <= 2; ++di)
#pragma unroll
        for (int dj = -2; dj <= 2; ++dj)
            m = fmaxf(m, a[ci + di * stride + dj]);
    return m;
}

// ---------------------------------------------------------------------------
// Kernel 0: zero histogram, candidate arrays, counters (ws is poisoned).
// ---------------------------------------------------------------------------
__global__ void dkd_init(unsigned int* __restrict__ hist,
                         unsigned long long* __restrict__ pairs,
                         unsigned int* __restrict__ cnt) {
    size_t t = (size_t)blockIdx.x * blockDim.x + threadIdx.x;
    size_t stride = (size_t)gridDim.x * blockDim.x;
    for (size_t i = t; i < (size_t)IMG_B * NBUCKET; i += stride) hist[i] = 0u;
    for (size_t i = t; i < (size_t)IMG_B * CAP2; i += stride) pairs[i] = 0ull;
    if (t < IMG_B) cnt[t] = 0u;
}

// ---------------------------------------------------------------------------
// Kernel 1: fused simple_nms (1 + 2 suppression iterations) + border zeroing.
// 32x32 output tile, halo 10 -> 52x52 LDS tile; shrinking mask fields in LDS.
// Interior tiles stream the score tile via GLOBAL_LOAD_ASYNC_TO_LDS_B32
// (ASYNCcnt); border tiles use the predicated fallback (-inf outside image).
// ---------------------------------------------------------------------------
__global__ __launch_bounds__(256) void dkd_nms(const float* __restrict__ scores,
                                               float* __restrict__ nms) {
    const int gx0 = blockIdx.x * 32, gy0 = blockIdx.y * 32, bimg = blockIdx.z;
    const float* img = scores + (size_t)bimg * IMG_N;
    const int tid = threadIdx.x;

    __shared__ float S[52 * 52];    // scores, radius 10
    __shared__ float M0[48 * 48];   // first max mask, radius 8
    __shared__ float SS1[44 * 44];  // supp_scores iter1, radius 6
    __shared__ float M1[40 * 40];   // mask after iter1, radius 4
    __shared__ float SS2[36 * 36];  // supp_scores iter2, radius 2

#if USE_ASYNC_LDS
    if (gx0 >= 10 && gy0 >= 10 && gx0 + 42 <= IMG_W && gy0 + 42 <= IMG_H) {
        for (int t = tid; t < 52 * 52; t += 256) {
            int i = t / 52, j = t % 52;
            const float* gp = img + (size_t)(gy0 - 10 + i) * IMG_W + (gx0 - 10 + j);
            __builtin_amdgcn_global_load_async_to_lds_b32(
                (g_int*)(float*)gp,
                (l_int*)&S[t],
                /*offset=*/0, /*cpol=*/0);
        }
        __builtin_amdgcn_s_wait_asynccnt(0);
    } else
#endif
    {
        for (int t = tid; t < 52 * 52; t += 256) {
            int i = t / 52, j = t % 52;
            int gy = gy0 - 10 + i, gx = gx0 - 10 + j;
            bool in = (gy >= 0 && gy < IMG_H && gx >= 0 && gx < IMG_W);
            S[t] = in ? img[gy * IMG_W + gx] : -INFINITY;
        }
    }
    __syncthreads();

    for (int t = tid; t < 48 * 48; t += 256) {
        int i = t / 48, j = t % 48;
        int ci = (i + 2) * 52 + (j + 2);
        float c = S[ci];
        float mx = max5(S, 52, ci);
        int gy = gy0 - 8 + i, gx = gx0 - 8 + j;
        bool in = (gy >= 0 && gy < IMG_H && gx >= 0 && gx < IMG_W);
        M0[t] = (in && c == mx) ? 1.0f : 0.0f;
    }
    __syncthreads();

    for (int t = tid; t < 44 * 44; t += 256) {
        int i = t / 44, j = t % 44;
        bool supp = max5(M0, 48, (i + 2) * 48 + (j + 2)) > 0.0f;
        int gy = gy0 - 6 + i, gx = gx0 - 6 + j;
        bool in = (gy >= 0 && gy < IMG_H && gx >= 0 && gx < IMG_W);
        float sc = S[(i + 4) * 52 + (j + 4)];
        SS1[t] = in ? (supp ? 0.0f : sc) : -INFINITY;
    }
    __syncthreads();

    for (int t = tid; t < 40 * 40; t += 256) {
        int i = t / 40, j = t % 40;
        int ci = (i + 2) * 44 + (j + 2);
        bool nm = (SS1[ci] == max5(SS1, 44, ci));
        bool supp = max5(M0, 48, (i + 4) * 48 + (j + 4)) > 0.0f;
        float m0c = M0[(i + 4) * 48 + (j + 4)];
        int gy = gy0 - 4 + i, gx = gx0 - 4 + j;
        bool in = (gy >= 0 && gy < IMG_H && gx >= 0 && gx < IMG_W);
        M1[t] = (in && (m0c > 0.0f || (nm && !supp))) ? 1.0f : 0.0f;
    }
    __syncthreads();

    for (int t = tid; t < 36 * 36; t += 256) {
        int i = t / 36, j = t % 36;
        bool supp = max5(M1, 40, (i + 2) * 40 + (j + 2)) > 0.0f;
        int gy = gy0 - 2 + i, gx = gx0 - 2 + j;
        bool in = (gy >= 0 && gy < IMG_H && gx >= 0 && gx < IMG_W);
        float sc = S[(i + 8) * 52 + (j + 8)];
        SS2[t] = in ? (supp ? 0.0f : sc) : -INFINITY;
    }
    __syncthreads();

    for (int t = tid; t < 32 * 32; t += 256) {
        int i = t / 32, j = t % 32;
        int ci = (i + 2) * 36 + (j + 2);
        bool nm = (SS2[ci] == max5(SS2, 36, ci));
        bool supp = max5(M1, 40, (i + 4) * 40 + (j + 4)) > 0.0f;
        float m1c = M1[(i + 4) * 40 + (j + 4)];
        bool keep = (m1c > 0.0f) || (nm && !supp);
        int gy = gy0 + i, gx = gx0 + j;
        float val = keep ? S[(i + 10) * 52 + (j + 10)] : 0.0f;
        if (gy < 2 || gy >= IMG_H - 2 || gx < 2 || gx >= IMG_W - 2) val = 0.0f;
        nms[(size_t)bimg * IMG_N + gy * IMG_W + gx] = val;
    }
}

// ---------------------------------------------------------------------------
// Kernel 2: per-image histogram over top-16 bits of the f32 pattern (exact
// order-isomorphic binning for positive floats).
// ---------------------------------------------------------------------------
__global__ void dkd_hist(const float* __restrict__ nms,
                         unsigned int* __restrict__ hist) {
    size_t t0 = (size_t)blockIdx.x * blockDim.x + threadIdx.x;
    size_t stride = (size_t)gridDim.x * blockDim.x;
    for (size_t t = t0; t < (size_t)IMG_B * IMG_N; t += stride) {
        float v = nms[t];
        if (v > 0.0f) {
            unsigned b = (unsigned)(t >> 20);
            unsigned bucket = __float_as_uint(v) >> 16;   // < NBUCKET
            atomicAdd(&hist[(size_t)b * NBUCKET + bucket], 1u);
        }
    }
}

// ---------------------------------------------------------------------------
// Kernel 3: per-image threshold bucket T: count(bucket > T) < TOPK <=
// count(bucket >= T). One block/image; two-level scan.
// ---------------------------------------------------------------------------
__global__ __launch_bounds__(256) void dkd_thresh(const unsigned int* __restrict__ hist,
                                                  unsigned int* __restrict__ thr) {
    __shared__ unsigned int csum[256];
    const int b = blockIdx.x;
    const unsigned int* h = hist + (size_t)b * NBUCKET;
    unsigned s = 0;
    const int c0 = threadIdx.x * (NBUCKET / 256);
#pragma unroll 4
    for (int i = 0; i < NBUCKET / 256; ++i) s += h[c0 + i];
    csum[threadIdx.x] = s;
    __syncthreads();
    if (threadIdx.x == 0) {
        unsigned cum = 0;
        unsigned T = 0;
        for (int c = 255; c >= 0; --c) {
            if (cum + csum[c] >= TOPK) {
                for (int bkt = c * (NBUCKET / 256) + (NBUCKET / 256) - 1;
                     bkt >= c * (NBUCKET / 256); --bkt) {
                    unsigned hc = h[bkt];
                    if (cum + hc >= TOPK) { T = (unsigned)bkt; break; }
                    cum += hc;
                }
                break;
            }
            cum += csum[c];
        }
        thr[b] = T;   // total < TOPK -> T = 0 (collect everything)
    }
}

// ---------------------------------------------------------------------------
// Kernel 4: compact survivors with bucket >= T into per-image 64-bit keys.
// key = (f32bits << 32) | (~idx): descending sort => value desc, index asc,
// matching jax.lax.top_k tie-breaking. Append order is nondeterministic but
// the SET is deterministic; the sort restores a deterministic order.
// ---------------------------------------------------------------------------
__global__ void dkd_collect(const float* __restrict__ nms,
                            const unsigned int* __restrict__ thr,
                            unsigned long long* __restrict__ pairs,
                            unsigned int* __restrict__ cnt) {
    size_t t0 = (size_t)blockIdx.x * blockDim.x + threadIdx.x;
    size_t stride = (size_t)gridDim.x * blockDim.x;
    for (size_t t = t0; t < (size_t)IMG_B * IMG_N; t += stride) {
        float v = nms[t];
        if (v > 0.0f) {
            unsigned b = (unsigned)(t >> 20);
            unsigned bits = __float_as_uint(v);
            if ((bits >> 16) >= thr[b]) {
                unsigned idx = (unsigned)(t & (IMG_N - 1));
                unsigned long long key = ((unsigned long long)bits << 32) |
                                         (unsigned long long)(0xFFFFFFFFu - idx);
                unsigned pos = atomicAdd(&cnt[b], 1u);
                if (pos < CAP2) pairs[(size_t)b * CAP2 + pos] = key;
            }
        }
    }
}

// ---------------------------------------------------------------------------
// Kernel 5: per-image bitonic sort (descending) of CAP2 keys; one block/image.
// ---------------------------------------------------------------------------
__global__ __launch_bounds__(1024) void dkd_sort(unsigned long long* __restrict__ pairs) {
    unsigned long long* p = pairs + (size_t)blockIdx.x * CAP2;
    const unsigned tid = threadIdx.x;
    for (unsigned k = 2; k <= CAP2; k <<= 1) {
        for (unsigned j = k >> 1; j > 0; j >>= 1) {
            for (unsigned i = tid; i < CAP2; i += 1024) {
                unsigned ixj = i ^ j;
                if (ixj > i) {
                    unsigned long long x = p[i], y = p[ixj];
                    bool up = ((i & k) == 0);
                    if (up ? (x < y) : (x > y)) { p[i] = y; p[ixj] = x; }
                }
            }
            __threadfence_block();
            __syncthreads();
        }
    }
}

// ---------------------------------------------------------------------------
// Kernel 6: per-keypoint soft-argmax via WMMA + dispersity + bilinear sample.
// Each wave handles 16 keypoints. A[16x32]f16 = x_exp patches (25 taps, zero
// padded), B[32x16]f16 cols = [dx, dy, 1, dx^2+dy^2]. One
// v_wmma_f32_16x16x32_f16 yields Sx,Sy,S1,S2 per keypoint.
//   rx = Sx/S1, ry = Sy/S1
//   dispersity = (S2 - (Sx^2+Sy^2)/S1) / (S1 * r^2)     (r = 2)
// ---------------------------------------------------------------------------
__global__ __launch_bounds__(256) void dkd_keypoints(const float* __restrict__ scores,
                                                     const unsigned long long* __restrict__ sorted,
                                                     float* __restrict__ out) {
    __shared__ _Float16 xe[8][16][32];   // [wave][keypoint][K]
    __shared__ float    Ssum[8][16][4];  // [wave][keypoint][Sx,Sy,S1,S2]

    const int tid = threadIdx.x;
    const int w = tid >> 5, lane = tid & 31;
    const int blk = blockIdx.x;          // 256 blocks; 32 per image
    const int b = blk >> 5;
    const int k0 = (blk & 31) * 128;
    const int m = lane & 15;
    const int kk = k0 + w * 16 + m;      // keypoint slot within image
    const float* img = scores + (size_t)b * IMG_N;

    int row = 0, col = 0;
    if (lane < 16) {
        unsigned long long key = sorted[(size_t)b * CAP2 + kk];
        unsigned idx = 0xFFFFFFFFu - (unsigned)(key & 0xFFFFFFFFull);
        if ((key >> 32) == 0ull) idx = 0u;   // degenerate padding guard
        row = (int)(idx >> 10);
        col = (int)(idx & (IMG_W - 1));
        float patch[25];
        float mx = -INFINITY;
#pragma unroll
        for (int i = 0; i < 5; ++i)
#pragma unroll
            for (int j = 0; j < 5; ++j) {
                int yy = row - 2 + i, xx = col - 2 + j;
                float v = (yy >= 0 && yy < IMG_H && xx >= 0 && xx < IMG_W)
                              ? img[yy * IMG_W + xx] : 0.0f;   // zero pad (Unfold)
                patch[i * 5 + j] = v;
                mx = fmaxf(mx, v);
            }
#pragma unroll
        for (int p = 0; p < 25; ++p)
            xe[w][m][p] = (_Float16)__expf((patch[p] - mx) * 10.0f); // /TEMPERATURE
#pragma unroll
        for (int p = 25; p < 32; ++p) xe[w][m][p] = (_Float16)0.0f;
    }
    __syncthreads();

    // A: 16-bit 16x32. lane<16: halves 0..15 -> K=0..15 (row=lane)
    //                  lane>=16: halves 0..15 -> K=16..31 (row=lane-16)
    v16h a;
    {
        const int r = lane & 15;
        const int kb0 = (lane < 16) ? 0 : 16;
#pragma unroll
        for (int h = 0; h < 16; ++h) {
            a[h]     = xe[w][r][kb0 + h];
        }
    }
    // B: 16-bit 32x16. lane n holds column (n&15); halves 0..15 -> K = base..base+15
    v16h bm;
    {
        const int ccol = lane & 15;
        const int kb = (lane < 16) ? 0 : 16;
#pragma unroll
        for (int h = 0; h < 16; ++h) {
            int k = kb + h;
            float v = 0.0f;
            if (k < 25 && ccol < 4) {
                float dx = (float)(k % 5 - 2);
                float dy = (float)(k / 5 - 2);
                v = (ccol == 0) ? dx : (ccol == 1) ? dy
                  : (ccol == 2) ? 1.0f : (dx * dx + dy * dy);
            }
            bm[h] = (_Float16)v;
        }
    }

    v8f c = {};
    // EXEC is all-ones here (divergent regions closed above) as WMMA requires.
    v8f d = __builtin_amdgcn_wmma_f32_16x16x32_f16(
        /*neg_a=*/false, a, /*neg_b=*/false, bm,
        /*c_mod=*/(short)0, c, /*reuse_a=*/false, /*reuse_b=*/false);

    // D: 16x16 f32. lane<16: D[v][lane]; lane>=16: D[8+v][lane-16]
    if ((lane & 15) < 4) {
        const int n = lane & 15;
        const int mb = (lane < 16) ? 0 : 8;
#pragma unroll
        for (int v = 0; v < 8; ++v) Ssum[w][mb + v][n] = d[v];
    }
    __syncthreads();

    if (lane < 16) {
        float Sx = Ssum[w][m][0], Sy = Ssum[w][m][1];
        float S1 = Ssum[w][m][2], S2 = Ssum[w][m][3];
        float inv = 1.0f / S1;
        float rx = Sx * inv, ry = Sy * inv;
        float disp = (S2 - (Sx * Sx + Sy * Sy) * inv) * inv * 0.25f;

        float kx = ((float)col + rx) / (float)(IMG_W - 1) * 2.0f - 1.0f;
        float ky = ((float)row + ry) / (float)(IMG_H - 1) * 2.0f - 1.0f;
        float px = (kx + 1.0f) * 0.5f * (float)(IMG_W - 1);
        float py = (ky + 1.0f) * 0.5f * (float)(IMG_H - 1);

        float x0 = floorf(px), y0 = floorf(py);
        float wx1 = px - x0, wx0 = 1.0f - wx1;
        float wy1 = py - y0, wy0 = 1.0f - wy1;
        auto g = [&](float xf, float yf) -> float {
            int xi = (int)xf; xi = xi < 0 ? 0 : (xi > IMG_W - 1 ? IMG_W - 1 : xi);
            int yi = (int)yf; yi = yi < 0 ? 0 : (yi > IMG_H - 1 ? IMG_H - 1 : yi);
            bool valid = (xf >= 0.0f) && (xf <= (float)(IMG_W - 1)) &&
                         (yf >= 0.0f) && (yf <= (float)(IMG_H - 1));
            return valid ? img[yi * IMG_W + xi] : 0.0f;
        };
        float score = wy0 * wx0 * g(x0, y0) + wy0 * wx1 * g(x0 + 1.0f, y0)
                    + wy1 * wx0 * g(x0, y0 + 1.0f) + wy1 * wx1 * g(x0 + 1.0f, y0 + 1.0f);

        size_t gk = (size_t)b * TOPK + (size_t)kk;
        out[gk * 2 + 0] = kx;                          // keypoints_xy
        out[gk * 2 + 1] = ky;
        out[(size_t)IMG_B * TOPK * 2 + gk] = disp;     // scoredispersity
        out[(size_t)IMG_B * TOPK * 3 + gk] = score;    // kptscores
    }
}

// ---------------------------------------------------------------------------
extern "C" void kernel_launch(void* const* d_in, const int* in_sizes, int n_in,
                              void* d_out, int out_size, void* d_ws, size_t ws_size,
                              hipStream_t stream) {
    (void)in_sizes; (void)n_in; (void)out_size; (void)ws_size;
    const float* scores = (const float*)d_in[0];
    float* out = (float*)d_out;

    // workspace layout (bytes)
    char* ws = (char*)d_ws;
    float* nms = (float*)ws;                                        // 32 MiB
    size_t off = (size_t)IMG_B * IMG_N * 4;
    unsigned int* hist = (unsigned int*)(ws + off);                 // 512 KiB
    off += (size_t)IMG_B * NBUCKET * 4;
    unsigned long long* pairs = (unsigned long long*)(ws + off);    // 512 KiB
    off += (size_t)IMG_B * CAP2 * 8;
    unsigned int* cnt = (unsigned int*)(ws + off);                  // 32 B
    off += 64;
    unsigned int* thr = (unsigned int*)(ws + off);                  // 32 B

    dkd_init<<<512, 256, 0, stream>>>(hist, pairs, cnt);
    dkd_nms<<<dim3(IMG_W / 32, IMG_H / 32, IMG_B), 256, 0, stream>>>(scores, nms);
    dkd_hist<<<4096, 256, 0, stream>>>(nms, hist);
    dkd_thresh<<<IMG_B, 256, 0, stream>>>(hist, thr);
    dkd_collect<<<4096, 256, 0, stream>>>(nms, thr, pairs, cnt);
    dkd_sort<<<IMG_B, 1024, 0, stream>>>(pairs);
    dkd_keypoints<<<(IMG_B * TOPK) / 128, 256, 0, stream>>>(scores, pairs, out);
}
